// EmbeddingCosineSimilarity_21199958573893
// MI455X (gfx1250) — compile-verified, matrix operand order checked by reading, not run on hardware
//
#include <hip/hip_runtime.h>

typedef __attribute__((ext_vector_type(16))) _Float16 v16h;
typedef __attribute__((ext_vector_type(8)))  _Float16 v8h;
typedef __attribute__((ext_vector_type(4)))  _Float16 v4h;
typedef __attribute__((ext_vector_type(8)))  float    v8f;

#define CLASSES 100000
#define CHANNEL 128
#define BATCH   4096

// ---------------------------------------------------------------------------
// Prep 1: L2-normalize each row of X (BATCH x CHANNEL, f32) -> Xn f16 row-major
// One wave (32 lanes) per row; each lane owns 4 consecutive floats.
// ---------------------------------------------------------------------------
__global__ __launch_bounds__(256) void xnorm_kernel(const float* __restrict__ X,
                                                    _Float16* __restrict__ Xn) {
  const int wave = (blockIdx.x * blockDim.x + threadIdx.x) >> 5;
  const int lane = threadIdx.x & 31;
  if (wave >= BATCH) return;

  const float4 v = ((const float4*)(X + (size_t)wave * CHANNEL))[lane];
  float s = v.x * v.x + v.y * v.y + v.z * v.z + v.w * v.w;
  // butterfly reduce across the 32 lanes (wave32)
  #pragma unroll
  for (int m = 16; m >= 1; m >>= 1) s += __shfl_xor(s, m, 32);
  const float inv = rsqrtf(fmaxf(s, 1e-12f));

  v4h o;
  o[0] = (_Float16)(v.x * inv);
  o[1] = (_Float16)(v.y * inv);
  o[2] = (_Float16)(v.z * inv);
  o[3] = (_Float16)(v.w * inv);
  ((v4h*)(Xn + (size_t)wave * CHANNEL))[lane] = o;
}

// ---------------------------------------------------------------------------
// Prep 2: L2-normalize each column k of W (CHANNEL x CLASSES, f32) and write
// the transposed result Wt (CLASSES x CHANNEL, f16 row-major).
// One thread per class column; column reads are coalesced across threads.
// W (f32) is dead after this kernel -> non-temporal loads so it does not
// evict the Wt panel we want resident in L2 for the GEMM.
// ---------------------------------------------------------------------------
__global__ __launch_bounds__(256) void wnorm_kernel(const float* __restrict__ W,
                                                    _Float16* __restrict__ Wt) {
  const int k = blockIdx.x * blockDim.x + threadIdx.x;
  if (k >= CLASSES) return;

  float s = 0.f;
  #pragma unroll 8
  for (int c = 0; c < CHANNEL; ++c) {
    const float v = __builtin_nontemporal_load(&W[(size_t)c * CLASSES + k]);
    s += v * v;
  }
  const float inv = rsqrtf(fmaxf(s, 1e-12f));
  #pragma unroll 8
  for (int c = 0; c < CHANNEL; ++c) {
    const float v = __builtin_nontemporal_load(&W[(size_t)c * CLASSES + k]);
    Wt[(size_t)k * CHANNEL + c] = (_Float16)(v * inv);
  }
}

// ---------------------------------------------------------------------------
// GEMM: out(B x CLASSES, f32) = Xn(B x C, f16) x Wt^T (Wt is CLASSES x C,
// f16 row-major). One wave computes a 64x32 output tile = 4x2 grid of 16x16
// v_wmma_f32_16x16x32_f16 tiles, K-loop of 4 chunks of 32 (32 WMMAs/wave).
// 4096/64 = 64 M-groups, 100000/32 = 3125 N-groups -> no bounds checks.
//
// Fragment layout per CDNA5 ISA (16-bit A 16x32): lanes 0-15 hold row (lane&15)
// with K = {0..7, 16..23}; lanes 16-31 hold the same rows with K = {8..15,
// 24..31}. Loaded as two B128 loads per fragment. B mirrors this from the
// transposed weights. C/D layout: VGPR j, lane L -> M = j + (L>=16?8:0),
// N = L&15.
//
// Output is streamed once (1.64 GB) -> non-temporal stores, so the streaming
// writes don't evict the 25.6 MB Wt panel from the 192 MB L2.
// ---------------------------------------------------------------------------
__global__ __launch_bounds__(256) void cosgemm_kernel(const _Float16* __restrict__ Xn,
                                                      const _Float16* __restrict__ Wt,
                                                      float* __restrict__ out) {
  const int lane = threadIdx.x & 31;
  const int wave = blockIdx.x * (blockDim.x >> 5) + (threadIdx.x >> 5);

  const int NT = CLASSES / 32;            // 3125 N-groups, exact
  const int mg = wave / NT;
  const int ng = wave - mg * NT;
  const int m0 = mg * 64;
  const int n0 = ng * 32;

  const int r       = lane & 15;
  const int halfSel = (lane & 16) ? 8 : 0;   // K sub-offset per ISA VGPR layout

  const _Float16* aBase = Xn + (size_t)(m0 + r) * CHANNEL + halfSel;
  const _Float16* bBase = Wt + (size_t)(n0 + r) * CHANNEL + halfSel;

  v8f acc[4][2];
  #pragma unroll
  for (int dm = 0; dm < 4; ++dm)
    #pragma unroll
    for (int dn = 0; dn < 2; ++dn)
      acc[dm][dn] = (v8f){};

  #pragma unroll
  for (int kc = 0; kc < 4; ++kc) {
    const int kb = kc * 32;

    union { v16h v; v8h h[2]; } a[4], b[2];
    #pragma unroll
    for (int dm = 0; dm < 4; ++dm) {
      const _Float16* p = aBase + (size_t)(dm * 16) * CHANNEL + kb;
      a[dm].h[0] = *(const v8h*)(p);
      a[dm].h[1] = *(const v8h*)(p + 16);
    }
    #pragma unroll
    for (int dn = 0; dn < 2; ++dn) {
      const _Float16* p = bBase + (size_t)(dn * 16) * CHANNEL + kb;
      b[dn].h[0] = *(const v8h*)(p);
      b[dn].h[1] = *(const v8h*)(p + 16);
    }

    #pragma unroll
    for (int dm = 0; dm < 4; ++dm)
      #pragma unroll
      for (int dn = 0; dn < 2; ++dn)
        acc[dm][dn] = __builtin_amdgcn_wmma_f32_16x16x32_f16(
            false, a[dm].v, false, b[dn].v,
            (short)0, acc[dm][dn], false, false);
  }

  const int col = n0 + r;
  #pragma unroll
  for (int dm = 0; dm < 4; ++dm) {
    const int rowBase = m0 + dm * 16 + ((lane & 16) ? 8 : 0);
    #pragma unroll
    for (int j = 0; j < 8; ++j) {
      __builtin_nontemporal_store(acc[dm][0][j],
          &out[(size_t)(rowBase + j) * CLASSES + col]);
      __builtin_nontemporal_store(acc[dm][1][j],
          &out[(size_t)(rowBase + j) * CLASSES + col + 16]);
    }
  }
}

// ---------------------------------------------------------------------------
// Workspace layout: [ Xn: BATCH*CHANNEL f16 | Wt: CLASSES*CHANNEL f16 ]
// = 1.0 MB + 25.6 MB = 26.6 MB total.
// ---------------------------------------------------------------------------
extern "C" void kernel_launch(void* const* d_in, const int* in_sizes, int n_in,
                              void* d_out, int out_size, void* d_ws, size_t ws_size,
                              hipStream_t stream) {
  (void)in_sizes; (void)n_in; (void)out_size; (void)ws_size;

  const float* X = (const float*)d_in[0];   // (BATCH, CHANNEL) f32
  const float* W = (const float*)d_in[1];   // (CHANNEL, CLASSES) f32
  float* out = (float*)d_out;               // (BATCH, CLASSES) f32

  _Float16* Xn = (_Float16*)d_ws;
  _Float16* Wt = Xn + (size_t)BATCH * CHANNEL;

  // Prep: normalize X rows (one wave per row, 8 waves per 256-thread block)
  xnorm_kernel<<<BATCH / 8, 256, 0, stream>>>(X, Xn);

  // Prep: normalize + transpose W columns
  wnorm_kernel<<<(CLASSES + 255) / 256, 256, 0, stream>>>(W, Wt);

  // GEMM: (4096/64) * (100000/32) = 64 * 3125 = 200000 wave-tiles,
  // 8 waves per block -> 25000 blocks.
  const int totalWaves = (BATCH / 64) * (CLASSES / 32);
  cosgemm_kernel<<<totalWaves / 8, 256, 0, stream>>>(Xn, Wt, out);
}
